// QLinearA16W4_86552180949215
// MI455X (gfx1250) — matrix operand order repaired and checked by reading, not compiled, not run on hardware
//
#include <hip/hip_runtime.h>

typedef __attribute__((ext_vector_type(16))) _Float16 v16h;
typedef __attribute__((ext_vector_type(8)))  _Float16 v8h;
typedef __attribute__((ext_vector_type(4)))  _Float16 v4h;
typedef __attribute__((ext_vector_type(8)))  float    v8f;

#define IN_F   4096
#define OUT_F  11008
#define TOKENS 512
#define GS     128

#define BM 128
#define BN 128
#define BK 128            // one quant group per K step
#define PITCH (BK + 8)    // 136 halves = 272 bytes: padded, 16B-aligned rows
#define KSTEPS (IN_F / BK)

#define XTILE (BM * PITCH)         // halves per X buffer
#define WTILE (BN * PITCH)         // halves per W buffer
#define WOFF  (2 * XTILE)          // W buffers start after both X buffers

union FragU { v16h v; v8h h[2]; };

// Stage one K-tile: x f32->f16 and w 4-bit->dequant f16 into LDS buffers.
__device__ __forceinline__
void stage_tiles(int kb, _Float16* __restrict__ Xb, _Float16* __restrict__ Wb,
                 const float* __restrict__ x, const int* __restrict__ qw,
                 const float* __restrict__ scales, const float* __restrict__ zeros,
                 int mbase, int nbase, int tid)
{
    const int xrow = tid >> 5;            // 0..7   (8 rows / pass, 16 passes)
    const int xc4  = tid & 31;            // float4 column
    const int wrow = tid >> 4;            // 0..15  (16 rows / pass, 8 passes)
    const int wc4  = tid & 15;            // int4 column (4 "bytes" -> 8 codes)

    // ---- X tile: f32 -> f16 ----
#pragma unroll
    for (int p = 0; p < 16; ++p) {
        const int row = p * 8 + xrow;
        const float4 vv =
            ((const float4*)(x + (size_t)(mbase + row) * IN_F + kb * BK))[xc4];
        v4h h;
        h.x = (_Float16)vv.x; h.y = (_Float16)vv.y;
        h.z = (_Float16)vv.z; h.w = (_Float16)vv.w;
        *(v4h*)&Xb[row * PITCH + xc4 * 4] = h;
    }

    // ---- W tile: unpack nibbles, group dequant ----
#pragma unroll
    for (int p = 0; p < 8; ++p) {
        const int n  = p * 16 + wrow;
        const int ng = nbase + n;
        const int g  = ng * (IN_F / GS) + kb;       // group id
        const float s = scales[g];
        const float z = zeros[g];

        const int* src = qw + (size_t)ng * (IN_F / 2) + kb * (BK / 2);
        const int4 b4  = ((const int4*)src)[wc4];
        if (kb + 1 < KSTEPS)
            __builtin_prefetch(src + (BK / 2) + wc4 * 4, 0, 0);   // next group

        const int by[4] = { b4.x, b4.y, b4.z, b4.w };
        _Float16 t[8];
#pragma unroll
        for (int j = 0; j < 4; ++j) {
            const float hf = (float)((by[j] >> 4) & 15);  // even code = hi nibble
            const float lf = (float)(by[j] & 15);         // odd  code = lo nibble
            t[2 * j]     = (_Float16)((hf - z) * s);
            t[2 * j + 1] = (_Float16)((lf - z) * s);
        }
        *(v8h*)&Wb[n * PITCH + wc4 * 8] = *(const v8h*)t;
    }
}

__global__ __launch_bounds__(256)
void qlinear_w4a16_wmma_kernel(const float* __restrict__ x,
                               const int*   __restrict__ qw,
                               const float* __restrict__ scales,
                               const float* __restrict__ zeros,
                               const float* __restrict__ bias,
                               float*       __restrict__ out)
{
    // dynamic LDS: double-buffered X and W tiles (139,264 bytes total)
    extern __shared__ _Float16 smem[];

    const int tid   = threadIdx.x;
    const int nbase = blockIdx.x * BN;    // output-feature tile
    const int mbase = blockIdx.y * BM;    // token tile

    const int lane = tid & 31;
    const int wave = tid >> 5;            // 0..7
    const int wm   = wave >> 2;           // 0..1  (64 rows of M each)
    const int wn   = wave & 3;            // 0..3  (32 cols of N each)
    const int r    = lane & 15;
    const int hi   = lane >> 4;           // lane half select

    v8f acc[4][2] = {};

    // prologue: stage first tile into buffer 0
    stage_tiles(0, smem, smem + WOFF, x, qw, scales, zeros, mbase, nbase, tid);
    __syncthreads();

#pragma unroll 1
    for (int kb = 0; kb < KSTEPS; ++kb) {
        // overlap: stage next tile into the other buffer while computing this one
        if (kb + 1 < KSTEPS) {
            const int pn = (kb + 1) & 1;
            stage_tiles(kb + 1,
                        smem + pn * XTILE,
                        smem + WOFF + pn * WTILE,
                        x, qw, scales, zeros, mbase, nbase, tid);
        }

        const int pc = kb & 1;
        const _Float16* __restrict__ Xb = smem + pc * XTILE;
        const _Float16* __restrict__ Wb = smem + WOFF + pc * WTILE;

        // ---- compute: 4 K-chunks of 32, 8 WMMAs each ----
#pragma unroll
        for (int kc = 0; kc < BK / 32; ++kc) {
            const int k0 = kc * 32;

            v16h a[4];
#pragma unroll
            for (int mt = 0; mt < 4; ++mt) {
                // A 16x32 f16: lanes 0-15 -> K 0..7 & 16..23; lanes 16-31 -> +8
                const int row = wm * 64 + mt * 16 + r;
                const int kA  = k0 + hi * 8;
                FragU u;
                u.h[0] = *(const v8h*)&Xb[row * PITCH + kA];
                u.h[1] = *(const v8h*)&Xb[row * PITCH + kA + 16];
                a[mt] = u.v;
            }

            v16h b[2];
#pragma unroll
            for (int nt = 0; nt < 2; ++nt) {
                // B 32x16 f16: lane = column n; lanes 0-15 hold K 0..7 & 16..23,
                // lanes 16-31 hold K 8..15 & 24..31 (same order as A)
                const int col = wn * 32 + nt * 16 + r;
                const int kB  = k0 + hi * 8;
                FragU u;
                u.h[0] = *(const v8h*)&Wb[col * PITCH + kB];
                u.h[1] = *(const v8h*)&Wb[col * PITCH + kB + 16];
                b[nt] = u.v;
            }

#pragma unroll
            for (int mt = 0; mt < 4; ++mt)
#pragma unroll
                for (int nt = 0; nt < 2; ++nt)
                    acc[mt][nt] = __builtin_amdgcn_wmma_f32_16x16x32_f16(
                        false, a[mt], false, b[nt],
                        (short)0, acc[mt][nt], false, false);
        }

        // single barrier per iteration:
        //  - compute(kb) reads done before buf kb&1 is overwritten at kb+2
        //  - stage(kb+1) writes visible before compute(kb+1)
        __syncthreads();
    }

    // ---- epilogue: C fragment layout -> global, add bias ----
#pragma unroll
    for (int mt = 0; mt < 4; ++mt) {
#pragma unroll
        for (int nt = 0; nt < 2; ++nt) {
            const int mg = mbase + wm * 64 + mt * 16 + hi * 8; // VGPR v -> row mg+v
            const int ng = nbase + wn * 32 + nt * 16 + r;      // lane  -> column
            const float bv = bias[ng];
#pragma unroll
            for (int v = 0; v < 8; ++v)
                out[(size_t)(mg + v) * OUT_F + ng] = acc[mt][nt][v] + bv;
        }
    }
}

extern "C" void kernel_launch(void* const* d_in, const int* in_sizes, int n_in,
                              void* d_out, int out_size, void* d_ws, size_t ws_size,
                              hipStream_t stream) {
    const float* x      = (const float*)d_in[0];
    const int*   qw     = (const int*)  d_in[1];
    const float* scales = (const float*)d_in[2];
    const float* zeros  = (const float*)d_in[3];
    const float* bias   = (const float*)d_in[4];
    float*       out    = (float*)d_out;

    const size_t shmem = (size_t)(2 * BM + 2 * BN) * PITCH * sizeof(_Float16); // 139,264 B
    dim3 grid(OUT_F / BN, TOKENS / BM);   // 86 x 4
    qlinear_w4a16_wmma_kernel<<<grid, 256, shmem, stream>>>(x, qw, scales, zeros, bias, out);
}
